// PonitaFiberBundle_77154792506114
// MI455X (gfx1250) — matrix operand order repaired and checked by reading, not compiled, not run on hardware
//
#include <hip/hip_runtime.h>
#include <hip/hip_bf16.h>
#include <math.h>

// ---------------- problem constants (match reference) ----------------
#define N_NODES 4096
#define N_EDGES 65536
#define O_ORI   12
#define HDIM    128
#define BDIM    64
#define IND     16
#define NLAYER  2
#define NGRAPH  8
#define WFH     512       // WF * H
#define RADIUS  3.0f

typedef __attribute__((ext_vector_type(16))) __bf16 v16bf;
typedef __attribute__((ext_vector_type(8)))  __bf16 v8bf;
typedef __attribute__((ext_vector_type(8)))  float  v8f;

// tanh via the hardware v_exp_f32 (exp2) transcendental: no libcalls.
__device__ __forceinline__ float tanh_hw(float t) {
    float e = __builtin_amdgcn_exp2f(t * 2.8853900817779268f);
    return 1.0f - 2.0f / (e + 1.0f);
}

__device__ __forceinline__ float gelu_f(float x) {
    // jax.nn.gelu (tanh approximation, the JAX default)
    return 0.5f * x * (1.0f + tanh_hw(0.7978845608028654f * (x + 0.044715f * x * x * x)));
}

__device__ __forceinline__ v8f wmma_bf16(v16bf a, v16bf b, v8f c) {
    return __builtin_amdgcn_wmma_f32_16x16x32_bf16(false, a, false, b, (short)0, c, false, false);
}

// B fragment from a pre-swizzled global buffer laid out [frag][lane(32)][j(16)] bf16.
__device__ __forceinline__ v16bf ldfrag_b(const __bf16* __restrict__ buf, int frag, int lane) {
    return *(const v16bf*)(buf + (((long)frag * 32 + lane) << 4));
}

// A fragment (16x32 bf16 sub-block) from a row-major [16][ldk] bf16 buffer.
__device__ __forceinline__ v16bf ldfrag_a(const __bf16* __restrict__ s, int m, int ldk, int kc, int kbase) {
    const __bf16* p = s + m * ldk + kc * 32 + kbase;
    v8bf lo = *(const v8bf*)(p);
    v8bf hi = *(const v8bf*)(p + 16);
    v16bf a;
#pragma unroll
    for (int i = 0; i < 8; i++) { a[i] = lo[i]; a[i + 8] = hi[i]; }
    return a;
}

__device__ __forceinline__ v8f v8f_zero() {
    v8f z = {0.f, 0.f, 0.f, 0.f, 0.f, 0.f, 0.f, 0.f};
    return z;
}

// ---------------- weight pre-swizzle: fp32 KxN -> bf16 B-fragment layout ----------------
__global__ void swizzle_w_kernel(const float* __restrict__ src, __bf16* __restrict__ dst,
                                 int Kreal, int Kpad, int Nn) {
    int i = blockIdx.x * blockDim.x + threadIdx.x;
    int total = Kpad * Nn;
    if (i >= total) return;
    int j    = i & 15;
    int lane = (i >> 4) & 31;
    int frag = i >> 9;
    int ntiles = Nn >> 4;
    int kc = frag / ntiles, nt = frag % ntiles;
    int k = kc * 32 + ((lane >> 4) << 3) + (j < 8 ? j : j + 8);   // table order: k = 8h + j (j<8), 16 + 8h + (j-8) (j>=8)
    int n = nt * 16 + (lane & 15);
    float v = (k < Kreal) ? src[(long)k * Nn + n] : 0.f;
    dst[i] = (__bf16)v;
}

// ---------------- CSR build over dst (so messages can be GATHERED, no float atomics) ----------------
__global__ void degree_kernel(const int* __restrict__ ei, int* __restrict__ deg) {
    int e = blockIdx.x * blockDim.x + threadIdx.x;
    if (e >= N_EDGES) return;
    atomicAdd(&deg[ei[N_EDGES + e]], 1);
}

__global__ void scan_kernel(const int* __restrict__ deg, int* __restrict__ rowstart) {
    if (blockIdx.x == 0 && threadIdx.x == 0) {      // 4096 ints: trivial serial scan
        int acc = 0;
        for (int n = 0; n < N_NODES; n++) { rowstart[n] = acc; acc += deg[n]; }
        rowstart[N_NODES] = acc;
    }
}

__global__ void scatter_kernel(const int* __restrict__ ei, const int* __restrict__ rowstart,
                               int* __restrict__ cursor, int* __restrict__ elist) {
    int e = blockIdx.x * blockDim.x + threadIdx.x;
    if (e >= N_EDGES) return;
    int d = ei[N_EDGES + e];
    int p = atomicAdd(&cursor[d], 1);
    elist[rowstart[d] + p] = e;
}

// ---------------- fb = mlp(ori@ori^T) and fbWfk[l] = fb @ Wfk_l  (tiny, VALU) ----------------
__global__ void fb_kernel(const float* __restrict__ ori,
                          const float* __restrict__ Wf1, const float* __restrict__ bf1,
                          const float* __restrict__ Wf2, const float* __restrict__ bf2,
                          const float* __restrict__ Wfk0, const float* __restrict__ Wfk1,
                          float* __restrict__ fbWfk /* [2][O*O][H] */) {
    __shared__ float s_fb[O_ORI * O_ORI * BDIM];
    int tid = threadIdx.x;
    if (tid < O_ORI * O_ORI) {
        int p = tid / O_ORI, o = tid % O_ORI;
        float d = ori[p * 3] * ori[o * 3] + ori[p * 3 + 1] * ori[o * 3 + 1] + ori[p * 3 + 2] * ori[o * 3 + 2];
        float f0 = d, f1 = d * d, f2 = f1 * d, f3 = f2 * d;   // poly_feats of scalar, degree 3
        float hbuf[HDIM];
        for (int h = 0; h < HDIM; h++) {
            float t = f0 * Wf1[0 * HDIM + h] + f1 * Wf1[1 * HDIM + h] +
                      f2 * Wf1[2 * HDIM + h] + f3 * Wf1[3 * HDIM + h] + bf1[h];
            hbuf[h] = gelu_f(t);
        }
        for (int b = 0; b < BDIM; b++) {
            float t = bf2[b];
            for (int h = 0; h < HDIM; h++) t += hbuf[h] * Wf2[h * BDIM + b];
            s_fb[tid * BDIM + b] = gelu_f(t);
        }
    }
    __syncthreads();
    int total = 2 * O_ORI * O_ORI * HDIM;
    for (int i = tid; i < total; i += blockDim.x) {
        int l  = i / (O_ORI * O_ORI * HDIM);
        int r  = i % (O_ORI * O_ORI * HDIM);
        int po = r / HDIM, c = r % HDIM;
        const float* Wfk = l ? Wfk1 : Wfk0;
        float t = 0.f;
        for (int b = 0; b < BDIM; b++) t += s_fb[po * BDIM + b] * Wfk[b * HDIM + c];
        fbWfk[i] = t;
    }
}

// ---------------- h = broadcast(x @ Wemb) over orientations ----------------
__global__ void embed_kernel(const float* __restrict__ x, const float* __restrict__ Wemb,
                             float* __restrict__ h) {
    int i = blockIdx.x * blockDim.x + threadIdx.x;   // n*H + c
    if (i >= N_NODES * HDIM) return;
    int n = i >> 7, c = i & 127;
    float t = 0.f;
#pragma unroll
    for (int k = 0; k < IND; k++) t += x[n * IND + k] * Wemb[k * HDIM + c];
#pragma unroll
    for (int o = 0; o < O_ORI; o++) h[((long)n * O_ORI + o) * HDIM + c] = t;
}

// ---------------- edge basis kernel: attr -> poly(30) -> 128 -> 64, *cutoff, bf16 kb ----------------
// One wave per 16-row M-tile of the flattened (e,o) space.  WMMA both GEMM stages.
__global__ void __launch_bounds__(256) edge_kb_kernel(
    const float* __restrict__ pos, const int* __restrict__ ei, const float* __restrict__ ori,
    const __bf16* __restrict__ Wb1s, const float* __restrict__ bb1,
    const __bf16* __restrict__ Wb2s, const float* __restrict__ bb2,
    __bf16* __restrict__ kb) {
    __shared__ __bf16 sMid[8][16 * HDIM];            // per-wave GELU'd mid activations (A for GEMM2)
    int lane = threadIdx.x & 31;
    int w    = threadIdx.x >> 5;
    int tile = blockIdx.x * 8 + w;                   // E*O/16 tiles, exact grid
    int m    = lane & 15;
    int hi   = lane >> 4;
    int kbase = hi ? 8 : 0;

    int row = tile * 16 + m;                         // flattened (e,o)
    int e = row / O_ORI, o = row % O_ORI;
    int src = ei[e], dst = ei[N_EDGES + e];
    float rx = pos[src * 3] - pos[dst * 3];
    float ry = pos[src * 3 + 1] - pos[dst * 3 + 1];
    float rz = pos[src * 3 + 2] - pos[dst * 3 + 2];
    float ox = ori[o * 3], oy = ori[o * 3 + 1], oz = ori[o * 3 + 2];
    float inv1 = rx * ox + ry * oy + rz * oz;
    float px = rx - inv1 * ox, py = ry - inv1 * oy, pz = rz - inv1 * oz;
    float inv2 = sqrtf(px * px + py * py + pz * pz);
    float dist = sqrtf(rx * rx + ry * ry + rz * rz);
    float tt = dist / RADIUS;
    float t3 = tt * tt * tt, t6 = t3 * t3;
    float cut = (1.f - 28.f * t6 + 48.f * t6 * tt - 21.f * t6 * tt * tt) * (dist < RADIUS ? 1.f : 0.f);

    // poly_feats(attr, 3): [a(2), a2(4), a3(8), a4(16)] = 30 features, padded to 32
    float f[32];
    f[0] = inv1; f[1] = inv2;
#pragma unroll
    for (int i = 0; i < 2; i++) { f[2 + 2 * i] = f[i] * inv1; f[3 + 2 * i] = f[i] * inv2; }
#pragma unroll
    for (int i = 0; i < 4; i++) { f[6 + 2 * i] = f[2 + i] * inv1; f[7 + 2 * i] = f[2 + i] * inv2; }
#pragma unroll
    for (int i = 0; i < 8; i++) { f[14 + 2 * i] = f[6 + i] * inv1; f[15 + 2 * i] = f[6 + i] * inv2; }
    f[30] = 0.f; f[31] = 0.f;

    // Build the A fragment (each lane redundantly computed its own row's features).
    v16bf af;
#pragma unroll
    for (int j = 0; j < 16; j++) {
        float vlo = (j < 8) ? f[j]     : f[8 + j];    // kbase = 0 half
        float vhi = (j < 8) ? f[8 + j] : f[16 + j];   // kbase = 8 half
        af[j] = (__bf16)(hi ? vhi : vlo);
    }

    // GEMM1: (16x32) @ (32x128), K fits one WMMA step.
#pragma unroll
    for (int nt = 0; nt < 8; nt++) {
        v8f acc = v8f_zero();
        acc = wmma_bf16(af, ldfrag_b(Wb1s, nt, lane), acc);
        int col = nt * 16 + m;
        float bias = bb1[col];
#pragma unroll
        for (int i = 0; i < 8; i++) {
            float v = gelu_f(acc[i] + bias);
            sMid[w][(i + hi * 8) * HDIM + col] = (__bf16)v;   // D -> A relayout via LDS
        }
    }
    // GEMM2: (16x128) @ (128x64)
#pragma unroll
    for (int nt = 0; nt < 4; nt++) {
        v8f acc = v8f_zero();
#pragma unroll
        for (int kc = 0; kc < 4; kc++) {
            v16bf a = ldfrag_a(&sMid[w][0], m, HDIM, kc, kbase);
            acc = wmma_bf16(a, ldfrag_b(Wb2s, kc * 4 + nt, lane), acc);
        }
        int col = nt * 16 + m;
        float bias = bb2[col];
#pragma unroll
        for (int i = 0; i < 8; i++) {
            float cw = __shfl(cut, i + hi * 8, 32);           // cutoff of output row
            float v  = gelu_f(acc[i] + bias) * cw;
            kb[((long)tile * 16 + i + hi * 8) * BDIM + col] = (__bf16)v;
        }
    }
}

// ---------------- message GATHER kernel ----------------
// x1[n] = sum over incoming edges e of (kb[e] @ Wk) * h[src[e]].
// One wave per (node, column-half): Wk B-fragments hoisted to registers,
// accumulation in registers, single store -- zero float atomics.
__global__ void __launch_bounds__(256) msg_gather_kernel(
    const __bf16* __restrict__ kb, const __bf16* __restrict__ Wks,
    const float* __restrict__ h, const int* __restrict__ ei,
    const int* __restrict__ rowstart, const int* __restrict__ elist,
    float* __restrict__ x1) {
    int lane = threadIdx.x & 31;
    int wid  = blockIdx.x * 8 + (threadIdx.x >> 5);  // 2*N_NODES waves, exact grid
    int n    = wid >> 1;
    int half = wid & 1;                              // columns [half*64, half*64+64)
    int m = lane & 15, hi = lane >> 4, kbase = hi ? 8 : 0;

    // Hoist the 8 Wk B-fragments this wave needs (re-used for every edge).
    v16bf B0[4], B1[4];
#pragma unroll
    for (int nt = 0; nt < 4; nt++) {
        B0[nt] = ldfrag_b(Wks, 0 * 8 + half * 4 + nt, lane);
        B1[nt] = ldfrag_b(Wks, 1 * 8 + half * 4 + nt, lane);
    }
    v8f acc[4];
#pragma unroll
    for (int i = 0; i < 4; i++) acc[i] = v8f_zero();

    int beg = rowstart[n], end = rowstart[n + 1];
    for (int q = beg; q < end; q++) {
        int e = elist[q];
        if (q + 1 < end)                             // prefetch next edge's kb row
            __builtin_prefetch(kb + ((long)elist[q + 1] * O_ORI + m) * BDIM, 0, 1);
        int src = ei[e];

        v16bf a0, a1;
#pragma unroll
        for (int j = 0; j < 16; j++) { a0[j] = (__bf16)0.f; a1[j] = (__bf16)0.f; }
        if (m < O_ORI) {
            const __bf16* base = kb + ((long)e * O_ORI + m) * BDIM + kbase;
            v8bf l0 = *(const v8bf*)(base);      v8bf h0 = *(const v8bf*)(base + 16);
            v8bf l1 = *(const v8bf*)(base + 32); v8bf h1 = *(const v8bf*)(base + 48);
#pragma unroll
            for (int j = 0; j < 8; j++) { a0[j] = l0[j]; a0[j + 8] = h0[j]; a1[j] = l1[j]; a1[j + 8] = h1[j]; }
        }
#pragma unroll
        for (int nt = 0; nt < 4; nt++) {
            v8f mid = v8f_zero();
            mid = wmma_bf16(a0, B0[nt], mid);
            mid = wmma_bf16(a1, B1[nt], mid);
            int col = (half * 4 + nt) * 16 + m;
#pragma unroll
            for (int i = 0; i < 8; i++) {
                int o = i + hi * 8;
                if (o < O_ORI)
                    acc[nt][i] += mid[i] * h[((long)src * O_ORI + o) * HDIM + col];
            }
        }
    }
    // write x1 (each (n, o, col) owned by exactly this wave; covers all rows -> no memset)
#pragma unroll
    for (int nt = 0; nt < 4; nt++) {
        int col = (half * 4 + nt) * 16 + m;
#pragma unroll
        for (int i = 0; i < 8; i++) {
            int o = i + hi * 8;
            if (o < O_ORI)
                x1[((long)n * O_ORI + o) * HDIM + col] = acc[nt][i];
        }
    }
}

// ---------------- fiber conv + LayerNorm + FFN (WMMA) + residual, per node ----------------
__global__ void __launch_bounds__(256) fiber_kernel(
    const float* __restrict__ x1, const float* __restrict__ fbWfk,
    const float* __restrict__ bc, const float* __restrict__ gW, const float* __restrict__ bW,
    const __bf16* __restrict__ W1s, const float* __restrict__ b1v,
    const __bf16* __restrict__ W2s, const float* __restrict__ b2v,
    float* __restrict__ h) {
    __shared__ __bf16 sX[8][16 * HDIM];   // normalized y (A of GEMM1)
    __shared__ __bf16 sM[8][16 * HDIM];   // GELU'd mid chunk (A of GEMM2)
    int lane = threadIdx.x & 31;
    int w    = threadIdx.x >> 5;
    int n    = blockIdx.x * 8 + w;        // one wave per node, exact grid
    int m = lane & 15, hi = lane >> 4, kbase = hi ? 8 : 0;

    // cache x1[n, :, c] for this lane's 4 channels
    float x1r[O_ORI][4];
#pragma unroll
    for (int o = 0; o < O_ORI; o++)
#pragma unroll
        for (int cc = 0; cc < 4; cc++)
            x1r[o][cc] = x1[((long)n * O_ORI + o) * HDIM + lane + 32 * cc];

    // zero pad rows 12..15 of sX
#pragma unroll
    for (int p = O_ORI; p < 16; p++)
#pragma unroll
        for (int cc = 0; cc < 4; cc++) sX[w][p * HDIM + lane + 32 * cc] = (__bf16)0.f;

    // x2 = einsum('oc,poc->pc', x1[n], fbWfk)/O + bc, then LayerNorm over c
    for (int p = 0; p < O_ORI; p++) {
        float v[4]; float s = 0.f, s2 = 0.f;
#pragma unroll
        for (int cc = 0; cc < 4; cc++) {
            int c = lane + 32 * cc;
            float t = 0.f;
#pragma unroll
            for (int o = 0; o < O_ORI; o++) t += x1r[o][cc] * fbWfk[(p * O_ORI + o) * HDIM + c];
            t = t * (1.f / O_ORI) + bc[c];
            v[cc] = t; s += t; s2 += t * t;
        }
#pragma unroll
        for (int d = 16; d >= 1; d >>= 1) { s += __shfl_xor(s, d, 32); s2 += __shfl_xor(s2, d, 32); }
        float mu = s * (1.f / HDIM);
        float var = s2 * (1.f / HDIM) - mu * mu;
        float rstd = rsqrtf(var + 1e-5f);
#pragma unroll
        for (int cc = 0; cc < 4; cc++) {
            int c = lane + 32 * cc;
            float y = (v[cc] - mu) * rstd * gW[c] + bW[c];
            sX[w][p * HDIM + c] = (__bf16)y;
        }
    }

    // FFN: gelu(y @ W1 + b1) @ W2 + b2, K-chunked so GEMM1 output feeds GEMM2 through LDS.
    v8f acc2[8];
#pragma unroll
    for (int i = 0; i < 8; i++) acc2[i] = v8f_zero();
    for (int cc = 0; cc < 4; cc++) {                 // 128-wide chunk of the 512 dim
#pragma unroll
        for (int nt = 0; nt < 8; nt++) {
            v8f mid = v8f_zero();
#pragma unroll
            for (int kc = 0; kc < 4; kc++) {
                v16bf a = ldfrag_a(&sX[w][0], m, HDIM, kc, kbase);
                mid = wmma_bf16(a, ldfrag_b(W1s, kc * 32 + cc * 8 + nt, lane), mid);
            }
            int col = cc * 128 + nt * 16 + m;
            float bias = b1v[col];
#pragma unroll
            for (int i = 0; i < 8; i++)
                sM[w][(i + hi * 8) * HDIM + nt * 16 + m] = (__bf16)gelu_f(mid[i] + bias);
        }
#pragma unroll
        for (int nt2 = 0; nt2 < 8; nt2++) {
#pragma unroll
            for (int kc2 = 0; kc2 < 4; kc2++) {
                v16bf a = ldfrag_a(&sM[w][0], m, HDIM, kc2, kbase);
                acc2[nt2] = wmma_bf16(a, ldfrag_b(W2s, (cc * 4 + kc2) * 8 + nt2, lane), acc2[nt2]);
            }
        }
    }
    // epilogue: h = (ffn_out + b2) + residual(h)
#pragma unroll
    for (int nt2 = 0; nt2 < 8; nt2++) {
        int col = nt2 * 16 + m;
        float bias = b2v[col];
#pragma unroll
        for (int i = 0; i < 8; i++) {
            int p = i + hi * 8;
            if (p < O_ORI) {
                long idx = ((long)n * O_ORI + p) * HDIM + col;
                h[idx] = h[idx] + acc2[nt2][i] + bias;
            }
        }
    }
}

// ---------------- readout accumulation: roAcc += h @ Wr + br ----------------
__global__ void readout_kernel(const float* __restrict__ h, const float* __restrict__ Wr,
                               const float* __restrict__ br, float* __restrict__ roAcc) {
    int i = blockIdx.x * blockDim.x + threadIdx.x;   // (n*O + o)
    if (i >= N_NODES * O_ORI) return;
    const float* hr = h + (long)i * HDIM;
    float acc[4] = {br[0], br[1], br[2], br[3]};
    for (int c = 0; c < HDIM; c++) {
        float hv = hr[c];
#pragma unroll
        for (int d = 0; d < 4; d++) acc[d] += hv * Wr[c * 4 + d];
    }
#pragma unroll
    for (int d = 0; d < 4; d++) roAcc[(long)i * 4 + d] += acc[d];
}

// ---------------- finalize the four outputs ----------------
__global__ void finalize_kernel(const float* __restrict__ roAcc, const float* __restrict__ ori,
                                const int* __restrict__ batch, float* __restrict__ out) {
    int n = blockIdx.x * blockDim.x + threadIdx.x;
    if (n >= N_NODES) return;
    float s0 = 0.f, s3 = 0.f, v1[3] = {0.f, 0.f, 0.f}, v2[3] = {0.f, 0.f, 0.f};
    for (int o = 0; o < O_ORI; o++) {
        const float* r = roAcc + ((long)n * O_ORI + o) * 4;
        s0 += r[0]; s3 += r[3];
#pragma unroll
        for (int d = 0; d < 3; d++) { v1[d] += r[1] * ori[o * 3 + d]; v2[d] += r[2] * ori[o * 3 + d]; }
    }
    const float sc = 1.f / (2.f * O_ORI);            // /len(outs) and /num_ori (or mean over O)
    out[n] = s0 * sc;
#pragma unroll
    for (int d = 0; d < 3; d++) out[N_NODES + n * 3 + d] = v1[d] * sc;
    int g = batch[n];
    atomicAdd(&out[4 * N_NODES + g], s3 * sc);
#pragma unroll
    for (int d = 0; d < 3; d++) atomicAdd(&out[4 * N_NODES + NGRAPH + g * 3 + d], v2[d] * sc);
}

// =====================================================================
extern "C" void kernel_launch(void* const* d_in, const int* in_sizes, int n_in,
                              void* d_out, int out_size, void* d_ws, size_t ws_size,
                              hipStream_t stream) {
    // Inputs in setup_inputs() dict insertion order (params leaves in insertion order):
    // 0 pos, 1 x, 2 ori_grid,
    // 3 Wb1, 4 bb1, 5 Wb2, 6 bb2, 7 Wf1, 8 bf1, 9 Wf2, 10 bf2, 11 Wemb,
    // layer0: 12 Wk, 13 Wfk, 14 bc, 15 g, 16 b, 17 W1, 18 b1, 19 W2, 20 b2
    // layer1: 21..29 (same order)
    // readout0: 30 W, 31 b; readout1: 32 W, 33 b
    // 34 edge_index, 35 batch
    const float* pos  = (const float*)d_in[0];
    const float* x    = (const float*)d_in[1];
    const float* ori  = (const float*)d_in[2];
    const float* Wb1  = (const float*)d_in[3];
    const float* bb1  = (const float*)d_in[4];
    const float* Wb2  = (const float*)d_in[5];
    const float* bb2  = (const float*)d_in[6];
    const float* Wf1  = (const float*)d_in[7];
    const float* bf1  = (const float*)d_in[8];
    const float* Wf2  = (const float*)d_in[9];
    const float* bf2  = (const float*)d_in[10];
    const float* Wemb = (const float*)d_in[11];
    const float* LP[2][9];
    for (int l = 0; l < 2; l++)
        for (int k = 0; k < 9; k++) LP[l][k] = (const float*)d_in[12 + l * 9 + k];
    const float* RW[2] = {(const float*)d_in[30], (const float*)d_in[32]};
    const float* RB[2] = {(const float*)d_in[31], (const float*)d_in[33]};
    const int* ei    = (const int*)d_in[34];
    const int* batch = (const int*)d_in[35];

    // ---- workspace carve-up ----
    char* ws = (char*)d_ws;
    size_t off = 0;
    auto alloc = [&](size_t bytes) -> void* {
        void* p = ws + off;
        off += (bytes + 255) & ~(size_t)255;
        return p;
    };
    __bf16* kb    = (__bf16*)alloc((size_t)N_EDGES * O_ORI * BDIM * 2);       // ~100 MB, L2-resident
    float*  h     = (float*)alloc((size_t)N_NODES * O_ORI * HDIM * 4);
    float*  x1    = (float*)alloc((size_t)N_NODES * O_ORI * HDIM * 4);
    float*  roAcc = (float*)alloc((size_t)N_NODES * O_ORI * 4 * 4);
    float*  fbWfk = (float*)alloc((size_t)2 * O_ORI * O_ORI * HDIM * 4);
    __bf16* Wb1s  = (__bf16*)alloc((size_t)32 * HDIM * 2);
    __bf16* Wb2s  = (__bf16*)alloc((size_t)HDIM * BDIM * 2);
    __bf16* Wks[2]; __bf16* W1s[2]; __bf16* W2s[2];
    for (int l = 0; l < 2; l++) {
        Wks[l] = (__bf16*)alloc((size_t)BDIM * HDIM * 2);
        W1s[l] = (__bf16*)alloc((size_t)HDIM * WFH * 2);
        W2s[l] = (__bf16*)alloc((size_t)WFH * HDIM * 2);
    }
    int* deg      = (int*)alloc((size_t)N_NODES * 4);
    int* rowstart = (int*)alloc((size_t)(N_NODES + 1) * 4);
    int* cursor   = (int*)alloc((size_t)N_NODES * 4);
    int* elist    = (int*)alloc((size_t)N_EDGES * 4);

    // ---- 1) weight pre-swizzle into WMMA B-fragment layout (bf16) ----
    auto swz = [&](const float* s, __bf16* d, int kr, int kp, int nn) {
        int tot = kp * nn;
        swizzle_w_kernel<<<(tot + 255) / 256, 256, 0, stream>>>(s, d, kr, kp, nn);
    };
    swz(Wb1, Wb1s, 30, 32, HDIM);
    swz(Wb2, Wb2s, HDIM, HDIM, BDIM);
    for (int l = 0; l < 2; l++) {
        swz(LP[l][0], Wks[l], BDIM, BDIM, HDIM);
        swz(LP[l][5], W1s[l], HDIM, HDIM, WFH);
        swz(LP[l][7], W2s[l], WFH, WFH, HDIM);
    }

    // ---- 2) CSR over dst (once; reused by both layers) ----
    (void)hipMemsetAsync(deg, 0, (size_t)N_NODES * 4, stream);
    (void)hipMemsetAsync(cursor, 0, (size_t)N_NODES * 4, stream);
    degree_kernel<<<N_EDGES / 256, 256, 0, stream>>>(ei, deg);
    scan_kernel<<<1, 1, 0, stream>>>(deg, rowstart);
    scatter_kernel<<<N_EDGES / 256, 256, 0, stream>>>(ei, rowstart, cursor, elist);

    // ---- 3) small precomputes ----
    (void)hipMemsetAsync(roAcc, 0, (size_t)N_NODES * O_ORI * 4 * 4, stream);
    fb_kernel<<<1, 256, 0, stream>>>(ori, Wf1, bf1, Wf2, bf2, LP[0][1], LP[1][1], fbWfk);
    embed_kernel<<<(N_NODES * HDIM + 255) / 256, 256, 0, stream>>>(x, Wemb, h);

    // ---- 4) edge basis kb (WMMA), shared by both layers ----
    edge_kb_kernel<<<(N_EDGES * O_ORI / 16) / 8, 256, 0, stream>>>(
        pos, ei, ori, Wb1s, bb1, Wb2s, bb2, kb);

    // ---- 5) two message-passing layers ----
    for (int l = 0; l < 2; l++) {
        msg_gather_kernel<<<(2 * N_NODES) / 8, 256, 0, stream>>>(
            kb, Wks[l], h, ei, rowstart, elist, x1);
        fiber_kernel<<<N_NODES / 8, 256, 0, stream>>>(
            x1, fbWfk + (size_t)l * O_ORI * O_ORI * HDIM,
            LP[l][2], LP[l][3], LP[l][4],
            W1s[l], LP[l][6], W2s[l], LP[l][8], h);
        readout_kernel<<<(N_NODES * O_ORI + 255) / 256, 256, 0, stream>>>(h, RW[l], RB[l], roAcc);
    }

    // ---- 6) finalize (zero graph-level accumulators first) ----
    (void)hipMemsetAsync((char*)d_out + (size_t)4 * N_NODES * 4, 0, (size_t)(NGRAPH + NGRAPH * 3) * 4, stream);
    finalize_kernel<<<(N_NODES + 255) / 256, 256, 0, stream>>>(roAcc, ori, batch, (float*)d_out);
}
